// VCTreeLSTMContext_60421599920153
// MI455X (gfx1250) — matrix operand order repaired and checked by reading, not compiled, not run on hardware
//
#include <hip/hip_runtime.h>
#include <hip/hip_bf16.h>
#include <math.h>

typedef unsigned short u16t;
typedef __attribute__((ext_vector_type(16))) __bf16 v16bf;
typedef __attribute__((ext_vector_type(8)))  float  v8f;
typedef __attribute__((ext_vector_type(8)))  unsigned short us8;
typedef __attribute__((ext_vector_type(16))) unsigned short us16v;

#define NN 4096
#define DD 4096
#define EE 200
#define HH 512
#define PP 128
#define CC 151
#define G4H 2048
#define K_OBJ 4448   /* 4424 padded to 32 */
#define K_DEC 4960   /* 4936 padded to 32 */
#define K_EDG 4832   /* 4808 padded to 32 */
#define K_EMB 224    /* 200  padded to 32 */
#define NWG_SCAN 16

__device__ __forceinline__ u16t f2bf(float x) {
  unsigned u = __float_as_uint(x);
  unsigned r = (u + 0x7FFFu + ((u >> 16) & 1u)) >> 16;
  return (u16t)r;
}
__device__ __forceinline__ float bf2f(u16t h) { return __uint_as_float(((unsigned)h) << 16); }
__device__ __forceinline__ float sigmoidf(float x) { return 1.0f / (1.0f + expf(-x)); }

union FragU { us16v u; v16bf v; };

// ---------------------------------------------------------------------------
// CDNA5 async global->LDS copy (ASYNCcnt-tracked).  lds_off is the low 32
// bits of the flat shared address (== wave-relative LDS byte address per the
// LDS aperture rule); gaddr is the 64-bit global address.
// ---------------------------------------------------------------------------
__device__ __forceinline__ void async_copy_b128(unsigned lds_off, const void* gaddr) {
  asm volatile("global_load_async_to_lds_b128 %0, %1, off"
               :: "v"(lds_off), "v"((unsigned long long)(uintptr_t)gaddr)
               : "memory");
}
__device__ __forceinline__ void wait_async0() {
  asm volatile("s_wait_asynccnt 0x0" ::: "memory");
}
__device__ __forceinline__ unsigned lds_off32(const void* p) {
  return (unsigned)(uintptr_t)p;
}

// ---------------------------------------------------------------------------
// Grid-wide barrier (16 co-resident workgroups), epoch-counter style.
// ---------------------------------------------------------------------------
__device__ void grid_barrier(int* bar, int nwg) {
  __syncthreads();
  if (threadIdx.x == 0) {
    int* cnt = bar;
    int* gen = bar + 1;
    __threadfence();
    int g = atomicAdd(gen, 0);
    if (atomicAdd(cnt, 1) == nwg - 1) {
      atomicExch(cnt, 0);
      __threadfence();
      atomicAdd(gen, 1);
    } else {
      while (atomicAdd(gen, 0) == g) { __builtin_amdgcn_s_sleep(1); }
    }
  }
  __syncthreads();
}

// ---------------------------------------------------------------------------
// WMMA bf16 GEMM: C[M,Nc] f32 = A[M,Kp] bf16 @ Bt^T (+ bias).
// Bt is the PRE-TRANSPOSED weight: Bt[n][k], n in [0,Nc), k in [0,Kp).
// Block: 256 threads (8 waves), tile 128(M) x 128(N), K step 32.
// Each wave owns a 16x128 strip: 1 A fragment, 8 B fragments, 8 WMMAs/step.
// Tiles staged via global_load_async_to_lds_b128.
// ---------------------------------------------------------------------------
__global__ __launch_bounds__(256) void wmma_gemm_bf16(
    const u16t* __restrict__ A, const u16t* __restrict__ Bt,
    const float* __restrict__ bias, float* __restrict__ C,
    int M, int Nc, int Kp)
{
  __shared__ __attribute__((aligned(16))) u16t Asm[128 * 32];
  __shared__ __attribute__((aligned(16))) u16t Bsm[128 * 32];   // Bsm[n][k]
  const int tid  = threadIdx.x;
  const int lane = tid & 31, wv = tid >> 5;
  const int rowBase = blockIdx.y * 128;
  const int colBase = blockIdx.x * 128;

  v8f acc[8];
  for (int i = 0; i < 8; ++i)
    for (int j = 0; j < 8; ++j) acc[i][j] = 0.0f;

  const int nk = Kp >> 5;
  const int ar = tid >> 1, aq = (tid & 1) << 4;  // 16 u16 (2 x b128) per thread
  const unsigned a_lds = lds_off32(Asm) + (unsigned)(ar * 32 + aq) * 2u;
  const unsigned b_lds = lds_off32(Bsm) + (unsigned)(ar * 32 + aq) * 2u;
  const bool a_ok = (rowBase + ar) < M;

  for (int kt = 0; kt < nk; ++kt) {
    // A tile: rows rowBase..+128, k cols kt*32..+32 (row-major)
    if (a_ok) {
      const u16t* src = A + (size_t)(rowBase + ar) * Kp + (kt << 5) + aq;
      async_copy_b128(a_lds, src);
      async_copy_b128(a_lds + 16, src + 8);
    } else {
      us8 z;
      for (int j = 0; j < 8; ++j) z[j] = 0;
      *(us8*)(Asm + ar * 32 + aq)     = z;
      *(us8*)(Asm + ar * 32 + aq + 8) = z;
    }
    // B tile: Bt rows colBase..+128, k cols kt*32..+32 (already transposed)
    {
      const u16t* src = Bt + (size_t)(colBase + ar) * Kp + (kt << 5) + aq;
      async_copy_b128(b_lds, src);
      async_copy_b128(b_lds + 16, src + 8);
    }
    wait_async0();
    __syncthreads();

    // A fragment: lane l -> M = wv*16 + (l&15); K runs [kh*8,+8) and [16+kh*8,+8)
    FragU fa;
    {
      int m = wv * 16 + (lane & 15), kh = lane >> 4;
      us8 lo = *(const us8*)(Asm + m * 32 + kh * 8);
      us8 hi = *(const us8*)(Asm + m * 32 + 16 + kh * 8);
      for (int j = 0; j < 8; ++j) { fa.u[j] = lo[j]; fa.u[8 + j] = hi[j]; }
    }
#pragma unroll
    for (int ns = 0; ns < 8; ++ns) {
      // B fragment: lane l -> N = ns*16 + (l&15); K runs [kh*8,+8) and [16+kh*8,+8)
      FragU fb;
      int n = ns * 16 + (lane & 15), kh = lane >> 4;
      us8 lo = *(const us8*)(Bsm + n * 32 + kh * 8);
      us8 hi = *(const us8*)(Bsm + n * 32 + 16 + kh * 8);
      for (int j = 0; j < 8; ++j) { fb.u[j] = lo[j]; fb.u[8 + j] = hi[j]; }
      acc[ns] = __builtin_amdgcn_wmma_f32_16x16x32_bf16(
          false, fa.v, false, fb.v, (short)0, acc[ns], false, false);
    }
    __syncthreads();
  }

  // Epilogue: lane l -> col (l&15); VGPR r -> row r + 8*(l>=16)
  for (int ns = 0; ns < 8; ++ns) {
    int n = colBase + ns * 16 + (lane & 15);
    float bv = bias ? bias[n] : 0.0f;
    int kh = lane >> 4;
    for (int r = 0; r < 8; ++r) {
      int m = rowBase + wv * 16 + r + 8 * kh;
      if (m < M) C[(size_t)m * Nc + n] = acc[ns][r] + bv;
    }
  }
}

// ---------------------------------------------------------------------------
// Build obj_pre (bf16, padded K_OBJ): [x | softmax(logits)@emb1 | posMLP | 0]
// ---------------------------------------------------------------------------
__global__ __launch_bounds__(256) void prep_obj_pre(
    const float* __restrict__ x, const float* __restrict__ logits,
    const float* __restrict__ boxes, const float* __restrict__ emb1,
    const float* __restrict__ pw1, const float* __restrict__ pb1,
    const float* __restrict__ pg, const float* __restrict__ pbeta,
    const float* __restrict__ pw2, const float* __restrict__ pb2,
    u16t* __restrict__ obj_pre)
{
  const int r = blockIdx.x, tid = threadIdx.x;
  __shared__ float lp[CC];
  __shared__ float h1[32];
  __shared__ float stats[1];
  if (tid < CC) lp[tid] = logits[(size_t)r * CC + tid];
  __syncthreads();
  if (tid == 0) {
    float m = lp[0];
    for (int i = 1; i < CC; ++i) m = fmaxf(m, lp[i]);
    float s = 0.0f;
    for (int i = 0; i < CC; ++i) { lp[i] = expf(lp[i] - m); s += lp[i]; }
    stats[0] = 1.0f / s;
  }
  if (tid < 32) {
    float s = pb1[tid];
    for (int k = 0; k < 9; ++k) s += boxes[(size_t)r * 9 + k] * pw1[k * 32 + tid];
    h1[tid] = s * pg[tid] + pbeta[tid];
  }
  __syncthreads();
  const float inv = stats[0];
  u16t* dst = obj_pre + (size_t)r * K_OBJ;
  for (int c = tid; c < K_OBJ; c += 256) {
    float v;
    if (c < DD) {
      v = x[(size_t)r * DD + c];
    } else if (c < DD + EE) {
      int e = c - DD;
      float s = 0.0f;
      for (int k = 0; k < CC; ++k) s += lp[k] * emb1[k * EE + e];
      v = s * inv;
    } else if (c < DD + EE + PP) {
      int p = c - DD - EE;
      float s = pb2[p];
      for (int k = 0; k < 32; ++k) s += h1[k] * pw2[k * PP + p];
      v = fmaxf(s, 0.0f);
    } else {
      v = 0.0f;
    }
    dst[c] = f2bf(v);
  }
}

// fp32 -> bf16 with zero padding (A-side matrices): dst[dstR][dstC]
__global__ void convert_pad(u16t* __restrict__ dst, const float* __restrict__ src,
                            int srcR, int srcC, int dstR, int dstC)
{
  size_t tot = (size_t)dstR * dstC;
  for (size_t i = (size_t)blockIdx.x * blockDim.x + threadIdx.x; i < tot;
       i += (size_t)gridDim.x * blockDim.x) {
    int rr = (int)(i / dstC), cc = (int)(i % dstC);
    float v = (rr < srcR && cc < srcC) ? src[(size_t)rr * srcC + cc] : 0.0f;
    dst[i] = f2bf(v);
  }
}

// fp32 [K][Nc] -> bf16 TRANSPOSED + K-padded: dst[Nc][Kp]
__global__ void convert_pad_t(u16t* __restrict__ dst, const float* __restrict__ src,
                              int K, int Nc, int Kp)
{
  size_t tot = (size_t)Nc * Kp;
  for (size_t i = (size_t)blockIdx.x * blockDim.x + threadIdx.x; i < tot;
       i += (size_t)gridDim.x * blockDim.x) {
    int n = (int)(i / Kp), k = (int)(i % Kp);
    float v = (k < K) ? src[(size_t)k * Nc + n] : 0.0f;
    dst[i] = f2bf(v);
  }
}

// d_out_w [512][151] fp32 -> outw_t [151][512] bf16
__global__ void conv_transpose_outw(u16t* __restrict__ dst, const float* __restrict__ src)
{
  int i = blockIdx.x * 256 + threadIdx.x;
  if (i < CC * HH) {
    int c = i / HH, r = i % HH;
    dst[i] = f2bf(src[(size_t)r * CC + c]);
  }
}

// hcat [N][1024] bf16 = [hf | hb]
__global__ void build_hcat(u16t* __restrict__ dst, const float* __restrict__ hf,
                           const float* __restrict__ hb)
{
  size_t tot = (size_t)NN * 1024;
  for (size_t i = (size_t)blockIdx.x * blockDim.x + threadIdx.x; i < tot;
       i += (size_t)gridDim.x * blockDim.x) {
    int r = (int)(i >> 10), c = (int)(i & 1023);
    float v = (c < HH) ? hf[(size_t)r * HH + c] : hb[(size_t)r * HH + (c - HH)];
    dst[i] = f2bf(v);
  }
}

// dec_inp [N][K_DEC] bf16 = [obj_pre(4424) | obj_ctx(512) | 0]
__global__ void build_dec_inp(u16t* __restrict__ dst, const u16t* __restrict__ obj_pre,
                              const float* __restrict__ ctx)
{
  size_t tot = (size_t)NN * K_DEC;
  for (size_t i = (size_t)blockIdx.x * blockDim.x + threadIdx.x; i < tot;
       i += (size_t)gridDim.x * blockDim.x) {
    int r = (int)(i / K_DEC), c = (int)(i % K_DEC);
    u16t v;
    if (c < DD + EE + PP)           v = obj_pre[(size_t)r * K_OBJ + c];
    else if (c < DD + EE + PP + HH) v = f2bf(ctx[(size_t)r * HH + (c - (DD + EE + PP))]);
    else                            v = 0;
    dst[i] = v;
  }
}

// edge_inp [N][K_EDG] bf16 = [emb2[pred](200) | x(4096) | obj_ctx(512) | 0]
__global__ void build_edge_inp(u16t* __restrict__ dst, const float* __restrict__ emb2,
                               const int* __restrict__ preds, const float* __restrict__ x,
                               const float* __restrict__ ctx)
{
  size_t tot = (size_t)NN * K_EDG;
  for (size_t i = (size_t)blockIdx.x * blockDim.x + threadIdx.x; i < tot;
       i += (size_t)gridDim.x * blockDim.x) {
    int r = (int)(i / K_EDG), c = (int)(i % K_EDG);
    float v;
    if (c < EE)                 v = emb2[(size_t)preds[r] * EE + c];
    else if (c < EE + DD)       v = x[(size_t)r * DD + (c - EE)];
    else if (c < EE + DD + HH)  v = ctx[(size_t)r * HH + (c - EE - DD)];
    else                        v = 0.0f;
    dst[i] = f2bf(v);
  }
}

// ---------------------------------------------------------------------------
// LSTM recurrence: 16 WGs, each holds a 512x128 bf16 slice of whh in LDS
// (gate cols {g*512 + wg*32 + jj}). Per step: 2KB h exchange via L2 (staged
// with async-to-LDS) + grid barrier.
// ---------------------------------------------------------------------------
__global__ __launch_bounds__(256) void lstm_scan(
    const float* __restrict__ pre, const u16t* __restrict__ whh,
    float* __restrict__ hs, float* hbuf /*2*512*/, int* bar, int nT, int rev)
{
  extern __shared__ __attribute__((aligned(16))) char smem[];
  u16t*  Wl  = (u16t*)smem;                   // 512*128 bf16 = 128KB
  float* hsm = (float*)(smem + 512 * 128 * 2);
  float* gv  = hsm + 512;
  const int tid = threadIdx.x, wg = blockIdx.x;
  const int j0 = wg * 32;
  for (int i = tid; i < 512 * 128; i += 256) {
    int k = i >> 7, c = i & 127;
    Wl[i] = whh[(size_t)k * G4H + ((c >> 5) << 9) + j0 + (c & 31)];
  }
  if (wg == 0)
    for (int i = tid; i < 1024; i += 256) hbuf[i] = 0.0f;
  float cst = 0.0f;
  __threadfence();
  grid_barrier(bar, NWG_SCAN);
  const int cidx = tid >> 1, hv = tid & 1;
  const unsigned hsm_lds = lds_off32(hsm) + (unsigned)tid * 16u;
  for (int s = 0; s < nT; ++s) {
    int t = rev ? (nT - 1 - s) : s;
    const float* hin = hbuf + (s & 1) * 512;
    float* hout = hbuf + ((s + 1) & 1) * 512;
    if (tid < 128) async_copy_b128(hsm_lds, hin + tid * 4);
    wait_async0();
    __syncthreads();
    float acc = 0.0f;
    int k0 = hv << 8;
#pragma unroll 8
    for (int k = 0; k < 256; ++k) acc += hsm[k0 + k] * bf2f(Wl[(k0 + k) * 128 + cidx]);
    acc += __shfl_xor(acc, 1, 32);
    if (hv == 0) {
      int gcol = ((cidx >> 5) << 9) + j0 + (cidx & 31);
      gv[cidx] = pre[(size_t)t * G4H + gcol] + acc;
    }
    __syncthreads();
    if (tid < 32) {
      float gi = gv[tid], gf = gv[32 + tid], gg = gv[64 + tid], go = gv[96 + tid];
      cst = sigmoidf(gf) * cst + sigmoidf(gi) * tanhf(gg);
      float h = sigmoidf(go) * tanhf(cst);
      hs[(size_t)t * HH + j0 + tid] = h;
      hout[j0 + tid] = h;
    }
    __threadfence();
    grid_barrier(bar, NWG_SCAN);
  }
}

// ---------------------------------------------------------------------------
// Decoder recurrence: same whh partition; WG0 also holds d_out_w^T (bf16,
// 155KB) in LDS for the per-step dist + argmax; prev-embedding term from the
// precomputed table eg[152][2048] indexed by previous argmax.
// ---------------------------------------------------------------------------
__global__ __launch_bounds__(256) void dec_scan(
    const float* __restrict__ pre_x, const u16t* __restrict__ whh,
    const float* __restrict__ eg, const u16t* __restrict__ outw_t,
    const float* __restrict__ out_b,
    float* __restrict__ dists, float* __restrict__ predsf, int* __restrict__ ipreds,
    float* hbuf /*2*512*/, int* predbuf, int* bar, int nT)
{
  extern __shared__ __attribute__((aligned(16))) char smem[];
  u16t*  Wl   = (u16t*)smem;
  float* hsm  = (float*)(smem + 512 * 128 * 2);
  float* gv   = hsm + 512;
  int*   spv  = (int*)(gv + 128);
  float* dsm  = (float*)(spv + 2);
  u16t*  OW   = (u16t*)(dsm + 152);
  const int tid = threadIdx.x, wg = blockIdx.x;
  const int j0 = wg * 32;
  for (int i = tid; i < 512 * 128; i += 256) {
    int k = i >> 7, c = i & 127;
    Wl[i] = whh[(size_t)k * G4H + ((c >> 5) << 9) + j0 + (c & 31)];
  }
  if (wg == 0) {
    for (int i = tid; i < CC * HH; i += 256) OW[i] = outw_t[i];
    for (int i = tid; i < 1024; i += 256) hbuf[i] = 0.0f;
  }
  float cst = 0.0f;
  __threadfence();
  grid_barrier(bar, NWG_SCAN);
  const int cidx = tid >> 1, hv = tid & 1;
  const unsigned hsm_lds = lds_off32(hsm) + (unsigned)tid * 16u;
  for (int t = 0; t < nT; ++t) {
    const float* hin = hbuf + (t & 1) * 512;
    float* hout = hbuf + ((t + 1) & 1) * 512;
    if (tid < 128) async_copy_b128(hsm_lds, hin + tid * 4);
    if (tid == 0) spv[0] = atomicAdd(predbuf, 0);
    wait_async0();
    __syncthreads();
    const int erow = spv[0] + 1;  // initial predbuf = -1 -> row 0 ('start')
    float acc = 0.0f;
    int k0 = hv << 8;
#pragma unroll 8
    for (int k = 0; k < 256; ++k) acc += hsm[k0 + k] * bf2f(Wl[(k0 + k) * 128 + cidx]);
    acc += __shfl_xor(acc, 1, 32);
    if (hv == 0) {
      int gcol = ((cidx >> 5) << 9) + j0 + (cidx & 31);
      gv[cidx] = pre_x[(size_t)t * G4H + gcol] + eg[(size_t)erow * G4H + gcol] + acc;
    }
    __syncthreads();
    if (tid < 32) {
      float gi = gv[tid], gf = gv[32 + tid], gg = gv[64 + tid], go = gv[96 + tid];
      cst = sigmoidf(gf) * cst + sigmoidf(gi) * tanhf(gg);
      float h = sigmoidf(go) * tanhf(cst);
      hout[j0 + tid] = h;
    }
    __threadfence();
    grid_barrier(bar, NWG_SCAN);
    if (wg == 0) {
      if (tid < 128) async_copy_b128(hsm_lds, hout + tid * 4);
      wait_async0();
      __syncthreads();
      if (tid < CC) {
        float s = out_b[tid];
        for (int k = 0; k < HH; ++k) s += hsm[k] * bf2f(OW[tid * HH + k]);
        dists[(size_t)t * CC + tid] = s;
        dsm[tid] = s;
      }
      __syncthreads();
      if (tid == 0) {
        int best = 0;
        float bv = dsm[0];
        for (int i = 1; i < CC; ++i)
          if (dsm[i] > bv) { bv = dsm[i]; best = i; }
        ipreds[t] = best;
        predsf[t] = (float)best;
        atomicExch(predbuf, best);
      }
    }
    __threadfence();
    grid_barrier(bar, NWG_SCAN);
  }
}

// ---------------------------------------------------------------------------
extern "C" void kernel_launch(void* const* d_in, const int* in_sizes, int n_in,
                              void* d_out, int out_size, void* d_ws, size_t ws_size,
                              hipStream_t stream)
{
  (void)in_sizes; (void)n_in; (void)out_size;
  const float* x       = (const float*)d_in[0];
  const float* logits  = (const float*)d_in[1];
  const float* boxes   = (const float*)d_in[2];
  const float* emb1    = (const float*)d_in[3];
  const float* pw1     = (const float*)d_in[4];
  const float* pb1     = (const float*)d_in[5];
  const float* pg      = (const float*)d_in[6];
  const float* pbeta   = (const float*)d_in[7];
  const float* pw2     = (const float*)d_in[8];
  const float* pb2     = (const float*)d_in[9];
  const float* o_wih_f = (const float*)d_in[10];
  const float* o_whh_f = (const float*)d_in[11];
  const float* o_b_f   = (const float*)d_in[12];
  const float* o_wih_b = (const float*)d_in[13];
  const float* o_whh_b = (const float*)d_in[14];
  const float* o_b_b   = (const float*)d_in[15];
  const float* o_proj  = (const float*)d_in[16];
  const float* o_pb    = (const float*)d_in[17];
  const float* dembW   = (const float*)d_in[18];
  const float* d_wih_x = (const float*)d_in[19];
  const float* d_wih_e = (const float*)d_in[20];
  const float* d_whhW  = (const float*)d_in[21];
  const float* d_bv    = (const float*)d_in[22];
  const float* d_outw  = (const float*)d_in[23];
  const float* d_outb  = (const float*)d_in[24];
  const float* emb2    = (const float*)d_in[25];
  const float* e_wih_f = (const float*)d_in[26];
  const float* e_whh_f = (const float*)d_in[27];
  const float* e_b_f   = (const float*)d_in[28];
  const float* e_wih_b = (const float*)d_in[29];
  const float* e_whh_b = (const float*)d_in[30];
  const float* e_b_b   = (const float*)d_in[31];
  const float* e_proj  = (const float*)d_in[32];
  const float* e_pb    = (const float*)d_in[33];

  char* ws = (char*)d_ws;
  size_t cur = 0;
  auto alloc = [&](size_t bytes) -> char* {
    char* p = ws + cur;
    cur = (cur + bytes + 255) & ~(size_t)255;
    return p;
  };

  u16t* objpre = (u16t*)alloc((size_t)NN * K_OBJ * 2);
  u16t* wAf    = (u16t*)alloc((size_t)G4H * K_OBJ * 2);   // transposed [N][Kp]
  u16t* wAb    = (u16t*)alloc((size_t)G4H * K_OBJ * 2);
  float* pre_f = (float*)alloc((size_t)NN * G4H * 4);
  float* pre_b = (float*)alloc((size_t)NN * G4H * 4);
  u16t* owhhF  = (u16t*)alloc((size_t)HH * G4H * 2);
  u16t* owhhB  = (u16t*)alloc((size_t)HH * G4H * 2);
  float* hs_f  = (float*)alloc((size_t)NN * HH * 4);
  float* hs_b  = (float*)alloc((size_t)NN * HH * 4);
  u16t* hcat   = (u16t*)alloc((size_t)NN * 1024 * 2);
  u16t* projW  = (u16t*)alloc((size_t)HH * 1024 * 2);     // transposed [512][1024]
  u16t* eprojW = (u16t*)alloc((size_t)HH * 1024 * 2);
  float* objctx= (float*)alloc((size_t)NN * HH * 4);
  u16t* decinp = (u16t*)alloc((size_t)NN * K_DEC * 2);
  u16t* dwx    = (u16t*)alloc((size_t)G4H * K_DEC * 2);   // transposed
  float* pre_x = (float*)alloc((size_t)NN * G4H * 4);
  u16t* demb   = (u16t*)alloc((size_t)(CC + 1) * K_EMB * 2 + 4096);
  u16t* dwe    = (u16t*)alloc((size_t)G4H * K_EMB * 2);   // transposed
  float* eg    = (float*)alloc((size_t)(CC + 1) * G4H * 4 + 4096);
  u16t* dwhh   = (u16t*)alloc((size_t)HH * G4H * 2);
  u16t* outwT  = (u16t*)alloc((size_t)CC * HH * 2);
  int*  ipreds = (int*)alloc((size_t)NN * 4);
  u16t* edginp = (u16t*)alloc((size_t)NN * K_EDG * 2);
  u16t* ewF    = (u16t*)alloc((size_t)G4H * K_EDG * 2);   // transposed
  u16t* ewB    = (u16t*)alloc((size_t)G4H * K_EDG * 2);
  u16t* ewhhF  = (u16t*)alloc((size_t)HH * G4H * 2);
  u16t* ewhhB  = (u16t*)alloc((size_t)HH * G4H * 2);
  float* hbuf  = (float*)alloc(2 * 512 * 4);
  int*  predbuf= (int*)alloc(256);
  int*  bar    = (int*)alloc(256);
  // reuse big buffers for edge phase (obj phase done by then)
  float* epre_f = pre_f;
  float* epre_b = pre_b;
  float* ehs_f  = hs_f;
  float* ehs_b  = hs_b;
  u16t*  ecat   = hcat;
  if (cur > ws_size) return;

  float* out_dists = (float*)d_out;
  float* out_preds = out_dists + (size_t)NN * CC;
  float* out_edge  = out_preds + NN;

  auto conv = [&](u16t* dst, const float* src, int sr, int sc, int dr, int dc) {
    size_t tot = (size_t)dr * dc;
    int blocks = (int)((tot + 255) / 256);
    if (blocks > 8192) blocks = 8192;
    convert_pad<<<blocks, 256, 0, stream>>>(dst, src, sr, sc, dr, dc);
  };
  auto convT = [&](u16t* dst, const float* src, int K, int Nc, int Kp) {
    size_t tot = (size_t)Nc * Kp;
    int blocks = (int)((tot + 255) / 256);
    if (blocks > 8192) blocks = 8192;
    convert_pad_t<<<blocks, 256, 0, stream>>>(dst, src, K, Nc, Kp);
  };
  auto gemm = [&](const u16t* A, const u16t* Bt, const float* bias, float* C,
                  int M, int Nc, int Kp) {
    dim3 g(Nc / 128, (M + 127) / 128);
    wmma_gemm_bf16<<<g, 256, 0, stream>>>(A, Bt, bias, C, M, Nc, Kp);
  };

  const size_t scan_lds = (size_t)512 * 128 * 2 + (512 + 128) * 4;
  const size_t dec_lds  = (size_t)512 * 128 * 2 + (512 + 128) * 4 + 2 * 4 + 152 * 4
                        + (size_t)CC * HH * 2;

  // --- phase 0: inputs -> bf16 (weights transposed once for the GEMMs) ---
  prep_obj_pre<<<NN, 256, 0, stream>>>(x, logits, boxes, emb1, pw1, pb1, pg, pbeta,
                                       pw2, pb2, objpre);
  convT(wAf, o_wih_f, DD + EE + PP, G4H, K_OBJ);
  convT(wAb, o_wih_b, DD + EE + PP, G4H, K_OBJ);
  conv(owhhF, o_whh_f, HH, G4H, HH, G4H);
  conv(owhhB, o_whh_b, HH, G4H, HH, G4H);
  convT(dwx, d_wih_x, DD + EE + PP + HH, G4H, K_DEC);
  conv(demb, dembW, CC + 1, EE, CC + 1, K_EMB);
  convT(dwe, d_wih_e, EE, G4H, K_EMB);
  conv(dwhh, d_whhW, HH, G4H, HH, G4H);
  convT(ewF, e_wih_f, EE + DD + HH, G4H, K_EDG);
  convT(ewB, e_wih_b, EE + DD + HH, G4H, K_EDG);
  conv(ewhhF, e_whh_f, HH, G4H, HH, G4H);
  conv(ewhhB, e_whh_b, HH, G4H, HH, G4H);
  convT(projW, o_proj, 2 * HH, HH, 2 * HH);
  convT(eprojW, e_proj, 2 * HH, HH, 2 * HH);
  conv_transpose_outw<<<(CC * HH + 255) / 256, 256, 0, stream>>>(outwT, d_outw);

  // --- phase 1: object context bi-LSTM ---
  gemm(objpre, wAf, o_b_f, pre_f, NN, G4H, K_OBJ);
  gemm(objpre, wAb, o_b_b, pre_b, NN, G4H, K_OBJ);
  hipMemsetAsync(bar, 0, 8, stream);
  lstm_scan<<<NWG_SCAN, 256, scan_lds, stream>>>(pre_f, owhhF, hs_f, hbuf, bar, NN, 0);
  hipMemsetAsync(bar, 0, 8, stream);
  lstm_scan<<<NWG_SCAN, 256, scan_lds, stream>>>(pre_b, owhhB, hs_b, hbuf, bar, NN, 1);
  build_hcat<<<4096, 256, 0, stream>>>(hcat, hs_f, hs_b);
  gemm(hcat, projW, o_pb, objctx, NN, HH, 2 * HH);

  // --- phase 2: decoder ---
  build_dec_inp<<<8192, 256, 0, stream>>>(decinp, objpre, objctx);
  gemm(decinp, dwx, d_bv, pre_x, NN, G4H, K_DEC);
  gemm(demb, dwe, nullptr, eg, CC + 1, G4H, K_EMB);
  hipMemsetAsync(bar, 0, 8, stream);
  hipMemsetAsync(predbuf, 0xFF, 4, stream);  // -1 -> 'start' row 0
  dec_scan<<<NWG_SCAN, 256, dec_lds, stream>>>(pre_x, dwhh, eg, outwT, d_outb,
                                               out_dists, out_preds, ipreds,
                                               hbuf, predbuf, bar, NN);

  // --- phase 3: edge context bi-LSTM ---
  build_edge_inp<<<8192, 256, 0, stream>>>(edginp, emb2, ipreds, x, objctx);
  gemm(edginp, ewF, e_b_f, epre_f, NN, G4H, K_EDG);
  gemm(edginp, ewB, e_b_b, epre_b, NN, G4H, K_EDG);
  hipMemsetAsync(bar, 0, 8, stream);
  lstm_scan<<<NWG_SCAN, 256, scan_lds, stream>>>(epre_f, ewhhF, ehs_f, hbuf, bar, NN, 0);
  hipMemsetAsync(bar, 0, 8, stream);
  lstm_scan<<<NWG_SCAN, 256, scan_lds, stream>>>(epre_b, ewhhB, ehs_b, hbuf, bar, NN, 1);
  build_hcat<<<4096, 256, 0, stream>>>(ecat, ehs_f, ehs_b);
  gemm(ecat, eprojW, e_pb, out_edge, NN, HH, 2 * HH);
}
